// GCN_9294309229069
// MI455X (gfx1250) — matrix operand order repaired and checked by reading, not compiled, not run on hardware
//
#include <hip/hip_runtime.h>
#include <hip/hip_bf16.h>
#include <stdint.h>

// ---------------------------------------------------------------------------
// GCN on MI455X (gfx1250, wave32).
//   - edge aggregation: L2-resident atomics (node array 25.6MB << 192MB L2)
//   - dense layers:     v_wmma_f32_16x16x32_f16, A from LDS, B from f16
//                       column-major weights, fused in-norm/bias/ReLU
// ---------------------------------------------------------------------------

#define N_NODES   100000
#define N_EDGES   1000000
#define FEATS     64

typedef __attribute__((ext_vector_type(16))) _Float16 v16h;
typedef __attribute__((ext_vector_type(8)))  float    v8f;

// ---------------------------- tiny utility kernels -------------------------

__global__ void gcn_zero_f32(float* __restrict__ p, int n) {
    int i = blockIdx.x * blockDim.x + threadIdx.x;
    if (i < n) p[i] = 0.0f;
}

__global__ void gcn_degrees(const int* __restrict__ src, const int* __restrict__ dst,
                            float* __restrict__ deg_out, float* __restrict__ deg_in, int n_edges) {
    int e = blockIdx.x * blockDim.x + threadIdx.x;
    if (e < n_edges) {
        atomicAdd(&deg_out[src[e]], 1.0f);
        atomicAdd(&deg_in [dst[e]], 1.0f);
    }
}

// deg -> rsqrt(max(deg,1))   (deg_out and deg_in are contiguous: 2N floats)
__global__ void gcn_rsqrt_clip(float* __restrict__ d, int n) {
    int i = blockIdx.x * blockDim.x + threadIdx.x;
    if (i < n) d[i] = rsqrtf(fmaxf(d[i], 1.0f));
}

// W (K x Nout, row-major f32) -> Wc (Nout x K, f16)  i.e. column-major,
// so the WMMA B-fragment (two runs of 8 consecutive K per lane) is a contiguous load.
__global__ void gcn_convert_w(const float* __restrict__ W, _Float16* __restrict__ Wc,
                              int K, int Nout) {
    int i = blockIdx.x * blockDim.x + threadIdx.x;
    if (i < K * Nout) {
        int k = i / Nout, n = i % Nout;
        Wc[n * K + k] = (_Float16)W[i];
    }
}

// ------------------------------- aggregation -------------------------------
// agg[dst] += X[src] * outnorm[src], one thread per (edge, feature).
// Feature rows are coalesced; index/outnorm loads are wave-broadcast and
// L2-resident, atomics land in L2 (node array fits easily).
__global__ void gcn_scatter_edges(const float* __restrict__ X,
                                  const float* __restrict__ outnorm,
                                  const int* __restrict__ src,
                                  const int* __restrict__ dst,
                                  float* __restrict__ agg, int n_edges) {
    unsigned long long t = (unsigned long long)blockIdx.x * blockDim.x + threadIdx.x;
    int e = (int)(t >> 6);
    if (e >= n_edges) return;
    int f = (int)(t & 63);
    int s = src[e], d = dst[e];
    float v = X[(size_t)s * FEATS + f] * outnorm[s];
    atomicAdd(&agg[(size_t)d * FEATS + f], v);
}

// --------------------------------- WMMA GEMM -------------------------------
// Y(N x n_out) = relu?( (rowscale .* X)(N x 64) @ W(64 x n_out) + bias )
// One block = one 16-row M tile; one wave per 16-col N tile (n_out/16 waves).
// K = 64 -> two v_wmma_f32_16x16x32_f16 per tile.
__global__ void gcn_gemm_wmma(const float* __restrict__ X,
                              const float* __restrict__ rowscale,   // may be null
                              const _Float16* __restrict__ Wc,      // n_out x 64 (col-major)
                              const float* __restrict__ bias,       // n_out
                              float* __restrict__ Y,
                              int n_out, int relu) {
    __shared__ _Float16 As[16][FEATS];     // 2 KB

    const int tid   = threadIdx.x;
    const int mbase = blockIdx.x * 16;

    // Stage A tile into LDS as f16, applying per-row in-norm scale.
    for (int idx = tid; idx < 16 * FEATS; idx += blockDim.x) {
        int r = idx >> 6, k = idx & 63;
        int row = mbase + r;
        float s = rowscale ? rowscale[row] : 1.0f;
        As[r][k] = (_Float16)(X[(size_t)row * FEATS + k] * s);
    }
    __syncthreads();

    const int wave = tid >> 5;
    const int lane = tid & 31;
    const int g    = lane >> 4;   // half-wave group
    const int r16  = lane & 15;   // M row (for A) / N col (for B)
    const int nb   = wave * 16;   // this wave's N-tile base

    v8f c = {};
#pragma unroll
    for (int kb = 0; kb < 2; ++kb) {
        const int Kb = kb * 32;
        // A fragment: 16-bit A 16x32 layout — lane holds K = Kb+g*8+i and
        // K = Kb+16+g*8+i (two contiguous 8-half runs).
        v16h a;
#pragma unroll
        for (int i = 0; i < 8; ++i) {
            a[i]     = As[r16][Kb + g * 8 + i];
            a[i + 8] = As[r16][Kb + 16 + g * 8 + i];
        }
        // B fragment: lane (n = r16, group g) holds K = Kb+g*8+h and
        // K = Kb+16+g*8+h (two contiguous 8-half runs), matching A.
        const _Float16* wp = Wc + (size_t)(nb + r16) * FEATS + Kb + g * 8;
        v16h b;
#pragma unroll
        for (int h = 0; h < 16; ++h) b[h] = wp[h < 8 ? h : h + 8];

        c = __builtin_amdgcn_wmma_f32_16x16x32_f16(
                /*neg_a=*/false, a, /*neg_b=*/false, b,
                /*c_mod=*/(short)0, c, /*reuse_a=*/false, /*reuse_b=*/false);
    }

    // C/D layout: VGPR r -> M = g*8 + r, N = r16. Fuse bias + ReLU.
    const int   col = nb + r16;
    const float bv  = bias[col];
#pragma unroll
    for (int r = 0; r < 8; ++r) {
        int row = mbase + g * 8 + r;
        float v = c[r] + bv;
        if (relu) v = fmaxf(v, 0.0f);
        Y[(size_t)row * n_out + col] = v;
    }
}

// --------------------------------- driver ----------------------------------

extern "C" void kernel_launch(void* const* d_in, const int* in_sizes, int n_in,
                              void* d_out, int out_size, void* d_ws, size_t ws_size,
                              hipStream_t stream) {
    const float* features = (const float*)d_in[0];
    const int*   src      = (const int*)  d_in[1];
    const int*   dst      = (const int*)  d_in[2];
    const float* W1       = (const float*)d_in[3];
    const float* b1       = (const float*)d_in[4];
    const float* W2       = (const float*)d_in[5];
    const float* b2       = (const float*)d_in[6];
    const float* Wm1      = (const float*)d_in[7];
    const float* bm1      = (const float*)d_in[8];
    const float* Wm2      = (const float*)d_in[9];
    const float* bm2      = (const float*)d_in[10];
    float* out = (float*)d_out;

    const int N = in_sizes[0] / FEATS;   // 100000
    const int E = in_sizes[1];           // 1000000

    // Workspace layout (all offsets 256B-aligned; total ~52.03 MB)
    char* ws = (char*)d_ws;
    float*    deg     = (float*)   (ws + 0);                       // 2N floats (out, then in)
    float*    deg_in  = deg + N;
    float*    bufA    = (float*)   (ws + 800000);                  // N*64 f32
    float*    bufB    = (float*)   (ws + 800000 + 25600000);       // N*64 f32
    _Float16* Wc1     = (_Float16*)(ws + 52000000);                // 64x64 f16
    _Float16* Wc2     = Wc1 + 64 * 64;
    _Float16* Wcm1    = Wc2 + 64 * 64;
    _Float16* Wcm2    = Wcm1 + 64 * 64;                            // 32x64 f16

    const int T = 256;
    const int nodeVec = N * FEATS;                 // 6.4M floats
    const int edgeThreads = 0;  (void)edgeThreads; (void)ws_size; (void)n_in; (void)out_size;

    // 1) degrees -> rsqrt(max(deg,1))
    gcn_zero_f32<<<(2 * N + T - 1) / T, T, 0, stream>>>(deg, 2 * N);
    gcn_degrees<<<(E + T - 1) / T, T, 0, stream>>>(src, dst, deg, deg_in, E);
    gcn_rsqrt_clip<<<(2 * N + T - 1) / T, T, 0, stream>>>(deg, 2 * N);

    // 2) weights -> f16 column-major (WMMA B-fragment friendly)
    gcn_convert_w<<<(64 * 64 + T - 1) / T, T, 0, stream>>>(W1,  Wc1,  64, 64);
    gcn_convert_w<<<(64 * 64 + T - 1) / T, T, 0, stream>>>(W2,  Wc2,  64, 64);
    gcn_convert_w<<<(64 * 64 + T - 1) / T, T, 0, stream>>>(Wm1, Wcm1, 64, 64);
    gcn_convert_w<<<(64 * 32 + T - 1) / T, T, 0, stream>>>(Wm2, Wcm2, 64, 32);

    const int mTiles     = N / 16;                                   // 6250 exact
    const unsigned long long sThreads = (unsigned long long)E * 64;  // 64M
    const int sBlocks    = (int)((sThreads + T - 1) / T);

    // 3) layer 1: agg(features * outnorm) -> GEMM(innorm, W1, b1, relu) -> bufB
    gcn_zero_f32<<<(nodeVec + T - 1) / T, T, 0, stream>>>(bufA, nodeVec);
    gcn_scatter_edges<<<sBlocks, T, 0, stream>>>(features, deg, src, dst, bufA, E);
    gcn_gemm_wmma<<<mTiles, 128, 0, stream>>>(bufA, deg_in, Wc1, b1, bufB, 64, 1);

    // 4) layer 2: agg(bufB * outnorm) -> GEMM(innorm, W2, b2, relu) -> bufB
    gcn_zero_f32<<<(nodeVec + T - 1) / T, T, 0, stream>>>(bufA, nodeVec);
    gcn_scatter_edges<<<sBlocks, T, 0, stream>>>(bufB, deg, src, dst, bufA, E);
    gcn_gemm_wmma<<<mTiles, 128, 0, stream>>>(bufA, deg_in, Wc2, b2, bufB, 64, 1);

    // 5) MLP head: relu(bufB @ Wm1 + bm1) -> bufA ; bufA @ Wm2 + bm2 -> out
    gcn_gemm_wmma<<<mTiles, 128, 0, stream>>>(bufB, nullptr, Wcm1, bm1, bufA, 64, 1);
    gcn_gemm_wmma<<<mTiles,  64, 0, stream>>>(bufA, nullptr, Wcm2, bm2, out,  32, 0);
}
